// GraspSamplerVAE_4002909519892
// MI455X (gfx1250) — matrix operand order repaired and checked by reading, not compiled, not run on hardware
//
#include <hip/hip_runtime.h>
#include <hip/hip_bf16.h>

// ---------------------------------------------------------------------------
// GraspSamplerVAE forward for MI455X (gfx1250, wave32, WMMA).
//
// GEMM precision: f16 A/B, f32 accumulate via V_WMMA_F32_16X16X32_F16.
// Each wave computes a 16(M) x 64(P) tile: one A fragment feeds 4 WMMAs per
// 32-deep k-step (4 f32 accumulators), raising FLOP/byte against L2/HBM.
// Conv/FC bias is skipped (cancels under train-mode BatchNorm).
// FPS / ball-query indices are shared between encoder and decoder (both
// depend only on `pc`).
// ---------------------------------------------------------------------------

typedef __attribute__((ext_vector_type(16))) _Float16 v16h;
typedef __attribute__((ext_vector_type(8)))  _Float16 v8h;
typedef __attribute__((ext_vector_type(8)))  float    v8f;

#define BQ 64      // batch
#define NPTS 1024  // points per cloud

static inline int rup32(int x) { return (x + 31) & ~31; }

// ---------------------------------------------------------------------------
// Weight prepack: f32 [M,K] row-major -> f16 [M,Kp] zero-padded.
// ---------------------------------------------------------------------------
__global__ void prepack_w_kernel(const float* __restrict__ W,
                                 _Float16* __restrict__ Wh,
                                 int M, int K, int Kp) {
  int t = blockIdx.x * blockDim.x + threadIdx.x;
  if (t >= M * Kp) return;
  int m = t / Kp, k = t % Kp;
  Wh[t] = (k < K) ? (_Float16)W[m * K + k] : (_Float16)0.0f;
}

// ---------------------------------------------------------------------------
// WMMA GEMM: Y[M,P] = Wh[M,Kp] * Xh[P,Kp]^T  (A=weights MxK, B=acts KxN).
// Block = 128 threads = 4 waves; each wave owns a 16x64 output tile
// (4 accumulators sharing one A fragment per k-step).
// grid.x = P/64, grid.y = M/64; wave w in block takes m-tile blockIdx.y*4+w.
// A fragment (16-bit 16x32): lanes 0-15 hold K 0..15, lanes 16-31
// hold K 16..31, M = lane&15.  B fragment (32x16): lanes 0-15 hold
// K 0..15, lanes 16-31 K 16..31, N = lane&15, contiguous per lane.
// ---------------------------------------------------------------------------
__global__ void gemm_wmma_kernel(const _Float16* __restrict__ A,  // [M,Kp]
                                 const _Float16* __restrict__ X,  // [P,Kp]
                                 float* __restrict__ Y,           // [M,P]
                                 int M, int Kp, int P) {
  int lid  = threadIdx.x & 31;
  int wave = threadIdx.x >> 5;
  int mt    = blockIdx.y * 4 + wave;  // 16-row tile index
  int pbase = blockIdx.x * 64;        // 64 columns per wave
  int m = mt * 16 + (lid & 15);
  int aoff = (lid < 16) ? 0 : 16;
  int boff = (lid < 16) ? 0 : 16;
  const _Float16* arow = A + (size_t)m * Kp + aoff;
  int prow = pbase + (lid & 15);
  const _Float16* b0 = X + (size_t)(prow +  0) * Kp + boff;
  const _Float16* b1 = X + (size_t)(prow + 16) * Kp + boff;
  const _Float16* b2 = X + (size_t)(prow + 32) * Kp + boff;
  const _Float16* b3 = X + (size_t)(prow + 48) * Kp + boff;
  v8f acc0 = {}, acc1 = {}, acc2 = {}, acc3 = {};
  for (int k0 = 0; k0 < Kp; k0 += 32) {
    v8h alo = *(const v8h*)(arow + k0);
    v8h ahi = *(const v8h*)(arow + k0 + 8);
    v16h av;
#pragma unroll
    for (int i = 0; i < 8; ++i) { av[i] = alo[i]; av[i + 8] = ahi[i]; }
    v16h bv0 = *(const v16h*)(b0 + k0);
    v16h bv1 = *(const v16h*)(b1 + k0);
    v16h bv2 = *(const v16h*)(b2 + k0);
    v16h bv3 = *(const v16h*)(b3 + k0);
    acc0 = __builtin_amdgcn_wmma_f32_16x16x32_f16(
        false, av, false, bv0, (short)0, acc0, false, false);
    acc1 = __builtin_amdgcn_wmma_f32_16x16x32_f16(
        false, av, false, bv1, (short)0, acc1, false, false);
    acc2 = __builtin_amdgcn_wmma_f32_16x16x32_f16(
        false, av, false, bv2, (short)0, acc2, false, false);
    acc3 = __builtin_amdgcn_wmma_f32_16x16x32_f16(
        false, av, false, bv3, (short)0, acc3, false, false);
  }
  int mbase = mt * 16 + ((lid < 16) ? 0 : 8);
  int pcol  = pbase + (lid & 15);
#pragma unroll
  for (int r = 0; r < 8; ++r) {
    size_t row = (size_t)(mbase + r) * P;
    Y[row + pcol +  0] = acc0[r];
    Y[row + pcol + 16] = acc1[r];
    Y[row + pcol + 32] = acc2[r];
    Y[row + pcol + 48] = acc3[r];
  }
}

// ---------------------------------------------------------------------------
// Per-channel BN statistics over all P points (one block per channel).
// ---------------------------------------------------------------------------
__global__ void colstats_kernel(const float* __restrict__ Y, int P,
                                float* __restrict__ mean,
                                float* __restrict__ rstd) {
  int m = blockIdx.x;
  const float* row = Y + (size_t)m * P;
  float s = 0.f, s2 = 0.f;
  for (int i = threadIdx.x; i < P; i += blockDim.x) {
    float v = row[i];
    s += v; s2 += v * v;
  }
  __shared__ float rs[256], rq[256];
  rs[threadIdx.x] = s; rq[threadIdx.x] = s2;
  __syncthreads();
  for (int st = 128; st > 0; st >>= 1) {
    if (threadIdx.x < st) {
      rs[threadIdx.x] += rs[threadIdx.x + st];
      rq[threadIdx.x] += rq[threadIdx.x + st];
    }
    __syncthreads();
  }
  if (threadIdx.x == 0) {
    float mu = rs[0] / (float)P;
    float var = rq[0] / (float)P - mu * mu;
    if (var < 0.f) var = 0.f;
    mean[m] = mu;
    rstd[m] = rsqrtf(var + 1e-5f);
  }
}

// ---------------------------------------------------------------------------
// BN + ReLU -> next-layer f16 activations [P,Kp] (zero padded).
// ---------------------------------------------------------------------------
__global__ void bn_relu_to_x_kernel(const float* __restrict__ Y,
                                    const float* __restrict__ mean,
                                    const float* __restrict__ rstd,
                                    const float* __restrict__ g,
                                    const float* __restrict__ be,
                                    _Float16* __restrict__ Xn,
                                    int M, int P, int Kp) {
  int p = blockIdx.x * blockDim.x + threadIdx.x;
  if (p >= P) return;
  _Float16* xp = Xn + (size_t)p * Kp;
  for (int m = 0; m < M; ++m) {
    float v = (Y[(size_t)m * P + p] - mean[m]) * rstd[m] * g[m] + be[m];
    xp[m] = (_Float16)(v > 0.f ? v : 0.f);
  }
  for (int m = M; m < Kp; ++m) xp[m] = (_Float16)0.0f;
}

// ---------------------------------------------------------------------------
// BN + ReLU + max over nsample -> pooled features F[cluster, M].
// ---------------------------------------------------------------------------
__global__ void bn_relu_maxpool_kernel(const float* __restrict__ Y,
                                       const float* __restrict__ mean,
                                       const float* __restrict__ rstd,
                                       const float* __restrict__ g,
                                       const float* __restrict__ be,
                                       float* __restrict__ F,
                                       int M, int P, int ns, int total) {
  int t = blockIdx.x * blockDim.x + threadIdx.x;
  if (t >= total) return;
  int c = t / M, m = t % M;
  const float* row = Y + (size_t)m * P + (size_t)c * ns;
  float mn = mean[m], rs = rstd[m], gg = g[m], bb = be[m];
  float best = -1e30f;
  for (int j = 0; j < ns; ++j) {
    float v = (row[j] - mn) * rs * gg + bb;
    v = v > 0.f ? v : 0.f;
    if (v > best) best = v;
  }
  F[(size_t)c * M + m] = best;
}

// ---------------------------------------------------------------------------
// Farthest point sampling, one block per batch.  Matches pointnet2 kernel:
// start at 0, dist init 1e10, argmax = first occurrence.
// ---------------------------------------------------------------------------
__global__ void fps_kernel(const float* __restrict__ xyz, int N, int npoint,
                           int* __restrict__ out) {
  __shared__ float dist[1024];
  __shared__ float rv[256];
  __shared__ int   ri[256];
  int b = blockIdx.x, tid = threadIdx.x;
  const float* px = xyz + (size_t)b * N * 3;
  for (int i = tid; i < N; i += 256) dist[i] = 1e10f;
  __syncthreads();
  int cur = 0;
  for (int it = 0; it < npoint; ++it) {
    if (tid == 0) out[b * npoint + it] = cur;
    float cx = px[cur * 3], cy = px[cur * 3 + 1], cz = px[cur * 3 + 2];
    float best = -1.f;
    int bi = 0x7fffffff;
    for (int i = tid; i < N; i += 256) {
      float dx = px[i * 3] - cx, dy = px[i * 3 + 1] - cy, dz = px[i * 3 + 2] - cz;
      float d = dx * dx + dy * dy + dz * dz;
      float nd = fminf(dist[i], d);
      dist[i] = nd;
      if (nd > best) { best = nd; bi = i; }
    }
    rv[tid] = best; ri[tid] = bi;
    __syncthreads();
    for (int st = 128; st > 0; st >>= 1) {
      if (tid < st) {
        if (rv[tid + st] > rv[tid] ||
            (rv[tid + st] == rv[tid] && ri[tid + st] < ri[tid])) {
          rv[tid] = rv[tid + st];
          ri[tid] = ri[tid + st];
        }
      }
      __syncthreads();
    }
    cur = ri[0];
    __syncthreads();
  }
}

__global__ void gather_xyz_kernel(const float* __restrict__ xyz,
                                  const int* __restrict__ fidx,
                                  float* __restrict__ nxyz,
                                  int N, int S, int total) {
  int t = blockIdx.x * blockDim.x + threadIdx.x;
  if (t >= total) return;
  int b = t / S;
  int id = fidx[t];
  const float* sp = xyz + ((size_t)b * N + id) * 3;
  nxyz[t * 3 + 0] = sp[0];
  nxyz[t * 3 + 1] = sp[1];
  nxyz[t * 3 + 2] = sp[2];
}

// Ball query with pointnet2 semantics: first ns in-radius indices, padded
// with the first in-radius index (0 if none).
__global__ void ball_query_kernel(const float* __restrict__ xyz,
                                  const float* __restrict__ nxyz,
                                  int* __restrict__ gidx,
                                  int N, int S, int ns, float r2, int total) {
  int t = blockIdx.x * blockDim.x + threadIdx.x;
  if (t >= total) return;
  int b = t / S;
  const float* px = xyz + (size_t)b * N * 3;
  float cx = nxyz[t * 3], cy = nxyz[t * 3 + 1], cz = nxyz[t * 3 + 2];
  int* out = gidx + (size_t)t * ns;
  int cnt = 0, first = 0;
  bool found = false;
  for (int n = 0; n < N && cnt < ns; ++n) {
    float dx = px[n * 3] - cx, dy = px[n * 3 + 1] - cy, dz = px[n * 3 + 2] - cz;
    float d = dx * dx + dy * dy + dz * dz;
    if (d <= r2) {
      if (!found) { found = true; first = n; }
      out[cnt++] = n;
    }
  }
  for (int j = cnt; j < ns; ++j) out[j] = first;
}

// ---------------------------------------------------------------------------
// Grouped-input builders (write f16 [P,Kp], zero padded).
// SA1 channels: centered xyz(3), pc(3), extra(grasp 16 | z 2), features(1).
// ---------------------------------------------------------------------------
__global__ void build_x_sa1_kernel(const float* __restrict__ pc,
                                   const float* __restrict__ extra, int extraC,
                                   const float* __restrict__ features,
                                   const float* __restrict__ nxyz,
                                   const int* __restrict__ gidx,
                                   _Float16* __restrict__ X,
                                   int N, int S, int ns, int Kp, int total) {
  int t = blockIdx.x * blockDim.x + threadIdx.x;
  if (t >= total) return;
  int bs = t / ns;
  int b = bs / S;
  int n = gidx[t];
  const float* pp = pc + ((size_t)b * N + n) * 3;
  const float* cc = nxyz + (size_t)bs * 3;
  _Float16* xp = X + (size_t)t * Kp;
  int c = 0;
  xp[c++] = (_Float16)(pp[0] - cc[0]);
  xp[c++] = (_Float16)(pp[1] - cc[1]);
  xp[c++] = (_Float16)(pp[2] - cc[2]);
  xp[c++] = (_Float16)pp[0];
  xp[c++] = (_Float16)pp[1];
  xp[c++] = (_Float16)pp[2];
  for (int e = 0; e < extraC; ++e) xp[c++] = (_Float16)extra[b * extraC + e];
  xp[c++] = (_Float16)features[(size_t)b * N + n];
  for (; c < Kp; ++c) xp[c] = (_Float16)0.0f;
}

// SA2 channels: centered xyz(3), F1 feats(C).
__global__ void build_x_sa2_kernel(const float* __restrict__ xyz1,  // [B,S1,3]
                                   const float* __restrict__ F1,    // [B*S1,C]
                                   const float* __restrict__ nxyz2, // [B*S2,3]
                                   const int* __restrict__ gidx2,
                                   _Float16* __restrict__ X,
                                   int S1, int S2, int ns, int C, int Kp,
                                   int total) {
  int t = blockIdx.x * blockDim.x + threadIdx.x;
  if (t >= total) return;
  int bs = t / ns;
  int b = bs / S2;
  int n = gidx2[t];
  const float* pp = xyz1 + ((size_t)b * S1 + n) * 3;
  const float* cc = nxyz2 + (size_t)bs * 3;
  const float* fp = F1 + ((size_t)b * S1 + n) * C;
  _Float16* xp = X + (size_t)t * Kp;
  xp[0] = (_Float16)(pp[0] - cc[0]);
  xp[1] = (_Float16)(pp[1] - cc[1]);
  xp[2] = (_Float16)(pp[2] - cc[2]);
  int c = 3;
  for (int k = 0; k < C; ++k) xp[c++] = (_Float16)fp[k];
  for (; c < Kp; ++c) xp[c] = (_Float16)0.0f;
}

// SA3 (group-all, no centering): xyz2(3) then F2 feats(C).
__global__ void build_x_sa3_kernel(const float* __restrict__ xyz2,  // [B*32,3]
                                   const float* __restrict__ F2,    // [B*32,C]
                                   _Float16* __restrict__ X,
                                   int C, int Kp, int total) {
  int t = blockIdx.x * blockDim.x + threadIdx.x;
  if (t >= total) return;
  const float* pp = xyz2 + (size_t)t * 3;
  const float* fp = F2 + (size_t)t * C;
  _Float16* xp = X + (size_t)t * Kp;
  xp[0] = (_Float16)pp[0];
  xp[1] = (_Float16)pp[1];
  xp[2] = (_Float16)pp[2];
  int c = 3;
  for (int k = 0; k < C; ++k) xp[c++] = (_Float16)fp[k];
  for (; c < Kp; ++c) xp[c] = (_Float16)0.0f;
}

// f32 [P,C] -> f16 [P,Kp] (FC input).
__global__ void f32_to_half_kernel(const float* __restrict__ F,
                                   _Float16* __restrict__ X,
                                   int P, int C, int Kp) {
  int p = blockIdx.x * blockDim.x + threadIdx.x;
  if (p >= P) return;
  _Float16* xp = X + (size_t)p * Kp;
  const float* fp = F + (size_t)p * C;
  int c = 0;
  for (; c < C; ++c) xp[c] = (_Float16)fp[c];
  for (; c < Kp; ++c) xp[c] = (_Float16)0.0f;
}

// ---------------------------------------------------------------------------
// Heads.
// ---------------------------------------------------------------------------
__global__ void heads_enc_kernel(const _Float16* __restrict__ h,  // [B,H]
                                 const float* __restrict__ Wmu,
                                 const float* __restrict__ bmu,
                                 const float* __restrict__ Wlv,
                                 const float* __restrict__ blv,
                                 const float* __restrict__ eps,
                                 float* __restrict__ mu_out,
                                 float* __restrict__ lv_out,
                                 float* __restrict__ z,
                                 int H, int L, int total) {
  int t = blockIdx.x * blockDim.x + threadIdx.x;
  if (t >= total) return;
  int b = t / L, o = t % L;
  const _Float16* hb = h + (size_t)b * H;
  float sm = 0.f, sl = 0.f;
  for (int k = 0; k < H; ++k) {
    float hv = (float)hb[k];
    sm += hv * Wmu[o * H + k];
    sl += hv * Wlv[o * H + k];
  }
  float mu = sm + bmu[o], lv = sl + blv[o];
  mu_out[t] = mu;
  lv_out[t] = lv;
  z[t] = mu + eps[t] * expf(0.5f * lv);
}

__global__ void heads_dec_kernel(const _Float16* __restrict__ h,  // [B,H]
                                 const float* __restrict__ Wq,
                                 const float* __restrict__ bq,
                                 const float* __restrict__ Wt,
                                 const float* __restrict__ bt,
                                 float* __restrict__ qt, int H) {
  int b = blockIdx.x * blockDim.x + threadIdx.x;
  if (b >= BQ) return;
  const _Float16* hb = h + (size_t)b * H;
  float q[4] = {0.f, 0.f, 0.f, 0.f};
  float tv[3] = {0.f, 0.f, 0.f};
  for (int k = 0; k < H; ++k) {
    float hv = (float)hb[k];
#pragma unroll
    for (int o = 0; o < 4; ++o) q[o] += hv * Wq[o * H + k];
#pragma unroll
    for (int o = 0; o < 3; ++o) tv[o] += hv * Wt[o * H + k];
  }
  float nrm = 0.f;
#pragma unroll
  for (int o = 0; o < 4; ++o) { q[o] += bq[o]; nrm += q[o] * q[o]; }
  nrm = fmaxf(sqrtf(nrm), 1e-12f);
#pragma unroll
  for (int o = 0; o < 4; ++o) qt[b * 7 + o] = q[o] / nrm;
#pragma unroll
  for (int o = 0; o < 3; ++o) qt[b * 7 + 4 + o] = tv[o] + bt[o];
}

// ---------------------------------------------------------------------------
// Host orchestration.
// ---------------------------------------------------------------------------
struct LayerP {
  const float *W, *b, *g, *be;
  int cin, cout;
};

extern "C" void kernel_launch(void* const* d_in, const int* in_sizes, int n_in,
                              void* d_out, int out_size, void* d_ws,
                              size_t ws_size, hipStream_t stream) {
  (void)in_sizes; (void)n_in; (void)out_size; (void)ws_size;
  const float* pc       = (const float*)d_in[0];  // [64,1024,3]
  const float* grasp    = (const float*)d_in[1];  // [64,16]
  const float* features = (const float*)d_in[2];  // [64,1024,1]
  const float* eps      = (const float*)d_in[3];  // [64,2]

  // Params flattened in setup_inputs() insertion order.
  int pi = 4;
  static const int cins_enc[11]  = {23, 64, 64, 131, 128, 128, 259, 256, 256, 512, 1024};
  static const int cins_dec[11]  = { 9, 64, 64, 131, 128, 128, 259, 256, 256, 512, 1024};
  static const int couts_[11]    = {64, 64, 128, 128, 128, 256, 256, 256, 512, 1024, 1024};
  LayerP enc[11], dec[11];
  for (int i = 0; i < 11; ++i) {
    enc[i] = {(const float*)d_in[pi], (const float*)d_in[pi + 1],
              (const float*)d_in[pi + 2], (const float*)d_in[pi + 3],
              cins_enc[i], couts_[i]};
    pi += 4;
  }
  for (int i = 0; i < 11; ++i) {
    dec[i] = {(const float*)d_in[pi], (const float*)d_in[pi + 1],
              (const float*)d_in[pi + 2], (const float*)d_in[pi + 3],
              cins_dec[i], couts_[i]};
    pi += 4;
  }
  const float* Wmu = (const float*)d_in[pi++];
  const float* bmu = (const float*)d_in[pi++];
  const float* Wlv = (const float*)d_in[pi++];
  const float* blv = (const float*)d_in[pi++];
  const float* Wq  = (const float*)d_in[pi++];
  const float* bq  = (const float*)d_in[pi++];
  const float* Wt  = (const float*)d_in[pi++];
  const float* bt  = (const float*)d_in[pi++];

  // Workspace bump allocator.
  char* base = (char*)d_ws;
  size_t off = 0;
  auto alloc = [&](size_t bytes) -> void* {
    off = (off + 255) & ~(size_t)255;
    void* p = base + off;
    off += bytes;
    return p;
  };
  const int S1 = 128, NS1 = 64, S2 = 32, NS2 = 128, NS3 = 32;
  const int P1 = BQ * S1 * NS1;   // 524288
  const int P2 = BQ * S2 * NS2;   // 262144
  const int P3 = BQ * NS3;        // 2048
  const int Pf = BQ;              // 64

  int*   fidx1 = (int*)alloc((size_t)BQ * S1 * 4);
  float* nxyz1 = (float*)alloc((size_t)BQ * S1 * 3 * 4);
  int*   gidx1 = (int*)alloc((size_t)BQ * S1 * NS1 * 4);
  int*   fidx2 = (int*)alloc((size_t)BQ * S2 * 4);
  float* nxyz2 = (float*)alloc((size_t)BQ * S2 * 3 * 4);
  int*   gidx2 = (int*)alloc((size_t)BQ * S2 * NS2 * 4);
  float* zbuf  = (float*)alloc((size_t)BQ * 2 * 4);
  float* F1 = (float*)alloc((size_t)BQ * S1 * 128 * 4);
  float* F2 = (float*)alloc((size_t)BQ * S2 * 256 * 4);
  float* F3 = (float*)alloc((size_t)BQ * 512 * 4);
  _Float16* Wh = (_Float16*)alloc((size_t)1024 * 1024 * 2);
  float* meanB = (float*)alloc(1024 * 4);
  float* rstdB = (float*)alloc(1024 * 4);
  size_t xbytes = (size_t)P2 * 160 * 2;  // 84 MB covers every activation shape
  _Float16* XA = (_Float16*)alloc(xbytes);
  _Float16* XB = (_Float16*)alloc(xbytes);
  float* Ybuf = (float*)alloc((size_t)128 * P1 * 4);  // 268 MB, max M*P

  auto cdiv = [](int a, int b) { return (a + b - 1) / b; };

  auto run_layer = [&](const LayerP& L, const _Float16* Xin, int P) {
    int Kp = rup32(L.cin);
    int tot = L.cout * Kp;
    prepack_w_kernel<<<cdiv(tot, 256), 256, 0, stream>>>(L.W, Wh, L.cout, L.cin, Kp);
    dim3 g(P / 64, L.cout / 64);  // 4 waves/block, one 16x64 tile per wave
    gemm_wmma_kernel<<<g, 128, 0, stream>>>(Wh, Xin, Ybuf, L.cout, Kp, P);
    colstats_kernel<<<L.cout, 256, 0, stream>>>(Ybuf, P, meanB, rstdB);
  };
  auto finish_x = [&](const LayerP& L, _Float16* Xout, int P) {
    int Kpo = rup32(L.cout);
    bn_relu_to_x_kernel<<<cdiv(P, 256), 256, 0, stream>>>(
        Ybuf, meanB, rstdB, L.g, L.be, Xout, L.cout, P, Kpo);
  };
  auto finish_pool = [&](const LayerP& L, float* F, int P, int ns) {
    int tot = (P / ns) * L.cout;
    bn_relu_maxpool_kernel<<<cdiv(tot, 256), 256, 0, stream>>>(
        Ybuf, meanB, rstdB, L.g, L.be, F, L.cout, P, ns, tot);
  };

  // --- shared geometry (depends only on pc) ---
  fps_kernel<<<BQ, 256, 0, stream>>>(pc, NPTS, S1, fidx1);
  gather_xyz_kernel<<<cdiv(BQ * S1, 256), 256, 0, stream>>>(pc, fidx1, nxyz1, NPTS, S1, BQ * S1);
  ball_query_kernel<<<cdiv(BQ * S1, 256), 256, 0, stream>>>(
      pc, nxyz1, gidx1, NPTS, S1, NS1, 0.02f * 0.02f, BQ * S1);
  fps_kernel<<<BQ, 256, 0, stream>>>(nxyz1, S1, S2, fidx2);
  gather_xyz_kernel<<<cdiv(BQ * S2, 256), 256, 0, stream>>>(nxyz1, fidx2, nxyz2, S1, S2, BQ * S2);
  ball_query_kernel<<<cdiv(BQ * S2, 256), 256, 0, stream>>>(
      nxyz1, nxyz2, gidx2, S1, S2, NS2, 0.04f * 0.04f, BQ * S2);

  auto run_backbone = [&](const LayerP* Lp, const float* extra, int extraC) -> _Float16* {
    build_x_sa1_kernel<<<cdiv(P1, 256), 256, 0, stream>>>(
        pc, extra, extraC, features, nxyz1, gidx1, XA, NPTS, S1, NS1, 32, P1);
    run_layer(Lp[0], XA, P1); finish_x(Lp[0], XB, P1);
    run_layer(Lp[1], XB, P1); finish_x(Lp[1], XA, P1);
    run_layer(Lp[2], XA, P1); finish_pool(Lp[2], F1, P1, NS1);

    build_x_sa2_kernel<<<cdiv(P2, 256), 256, 0, stream>>>(
        nxyz1, F1, nxyz2, gidx2, XA, S1, S2, NS2, 128, 160, P2);
    run_layer(Lp[3], XA, P2); finish_x(Lp[3], XB, P2);
    run_layer(Lp[4], XB, P2); finish_x(Lp[4], XA, P2);
    run_layer(Lp[5], XA, P2); finish_pool(Lp[5], F2, P2, NS2);

    build_x_sa3_kernel<<<cdiv(P3, 256), 256, 0, stream>>>(nxyz2, F2, XA, 256, 288, P3);
    run_layer(Lp[6], XA, P3); finish_x(Lp[6], XB, P3);
    run_layer(Lp[7], XB, P3); finish_x(Lp[7], XA, P3);
    run_layer(Lp[8], XA, P3); finish_pool(Lp[8], F3, P3, NS3);

    f32_to_half_kernel<<<1, 64, 0, stream>>>(F3, XA, Pf, 512, 512);
    run_layer(Lp[9], XA, Pf);  finish_x(Lp[9], XB, Pf);
    run_layer(Lp[10], XB, Pf); finish_x(Lp[10], XA, Pf);
    return XA;  // [64,1024] f16
  };

  // --- encoder -> mu/logvar/z ---
  _Float16* hEnc = run_backbone(enc, grasp, 16);
  float* out = (float*)d_out;
  heads_enc_kernel<<<1, 128, 0, stream>>>(hEnc, Wmu, bmu, Wlv, blv, eps,
                                          out + 448, out + 576, zbuf,
                                          1024, 2, BQ * 2);
  // --- decoder -> qt ---
  _Float16* hDec = run_backbone(dec, zbuf, 2);
  heads_dec_kernel<<<1, 64, 0, stream>>>(hDec, Wq, bq, Wt, bt, out, 1024);
}
